// ViT_78168404787325
// MI455X (gfx1250) — matrix operand order repaired and checked by reading, not compiled, not run on hardware
//
#include <hip/hip_runtime.h>

// ---------------------------------------------------------------------------
// ViT forward for gfx1250 (MI455X). All GEMMs run on v_wmma_f32_16x16x32_bf16
// (bf16 inputs via native float->bf16 conversion in LDS staging, f32
// accumulation; wave32 layouts per CDNA5 ISA 7.12.2). Every GEMM dimension is
// padded to tile multiples so the kernel is a single unguarded fast path.
// Each wave computes two 16x64 output strips, reusing B fragments: 8 WMMAs
// per K-step.
// ---------------------------------------------------------------------------

typedef __bf16 bf16;
typedef __attribute__((ext_vector_type(16))) __bf16 v16bf;
typedef __attribute__((ext_vector_type(8)))  __bf16 v8bf;
typedef __attribute__((ext_vector_type(2)))  __bf16 v2bf;
typedef __attribute__((ext_vector_type(8)))  float  v8f;
typedef __attribute__((ext_vector_type(2)))  float  v2f;

#define HIDN      384
#define HEADS_N   8
#define DH_N      64
#define INNER_N   512
#define MLP_N     1536
#define DEPTH_N   6
#define OUT_N     1000
#define OUT_PAD   1024
#define BATCH_N   16
#define LTOK      577          // 576 patches + cls appended at the end
#define L_PAD     640          // N/K pad for attention (10*64 / 20*32)
#define L_MPAD    768          // M pad for attention (3*256)
#define NPATCH    576
#define PDIM      768          // 16*16*3
#define MROWS     (BATCH_N * LTOK)   // 9232 real token rows
#define MROWS_PAD 9472               // 37*256 padded token rows

#define BM 256
#define BN 64
#define BK 32
#define LDS_PAD 8
#define LDS_STRIDE (BK + LDS_PAD)   // 40 bf16 = 80 bytes -> 16B aligned rows

#define EPI_NONE 0
#define EPI_GELU 1

// float -> bf16 conversions: let the backend select the native cvt ops.
__device__ __forceinline__ unsigned pack2(float lo, float hi) {
  v2f in; in.x = lo; in.y = hi;
  v2bf r = __builtin_convertvector(in, v2bf);
  return __builtin_bit_cast(unsigned, r);
}
__device__ __forceinline__ unsigned short bf1(float v) {
  bf16 b = (bf16)v;
  return __builtin_bit_cast(unsigned short, b);
}

// C[z] = epi(alpha * A[z] @ B[z] + bias) + Res[z]
// A: row-major (lda). B: NN => K x N (ldb) ; NT => N x K (ldb).
// Batch offset for matrix X: (z/div)*xOut + (z%div)*xIn.
// PRECONDITIONS (enforced by the launcher via padding):
//   gridDim.y*BM rows and gridDim.x*BN cols are fully in-bounds,
//   K % 32 == 0, lda/ldb/ldc and all offsets are multiples of 4 floats.
__global__ __launch_bounds__(256)
void gemm_wmma_bf16(const float* __restrict__ A, long long lda, long long aOut, long long aIn,
                    const float* __restrict__ Bm, long long ldb, long long bOut, long long bIn,
                    int bTrans,
                    const float* __restrict__ bias,
                    const float* __restrict__ Res, long long ldr, long long rOut, long long rIn,
                    float* __restrict__ C, long long ldc, long long cOut, long long cIn,
                    int K, int batchDiv, float alpha, int epi)
{
  __shared__ __align__(16) bf16 As[BM][LDS_STRIDE];   // 256 x 40 bf16 = 20 KB
  __shared__ __align__(16) bf16 Bs[BN][LDS_STRIDE];   //  64 x 40 bf16 =  5 KB

  const int z  = blockIdx.z;
  const long long zo = z / batchDiv, zi = z % batchDiv;
  A  += zo * aOut + zi * aIn;
  Bm += zo * bOut + zi * bIn;
  C  += zo * cOut + zi * cIn;
  if (Res) Res += zo * rOut + zi * rIn;

  const int m0   = blockIdx.y * BM;
  const int n0   = blockIdx.x * BN;
  const int tid  = threadIdx.x;
  const int lane = tid & 31;
  const int wave = tid >> 5;
  const int half = lane >> 4;          // 0: lanes 0-15, 1: lanes 16-31
  const int lmod = lane & 15;
  const int arow = wave * 16 + lmod;   // first local A row for this lane
  const int akoff = half * 8;          // A frag K offset per ISA layout

  // staging thread mappings (constant trip counts, vector loads)
  const int aM = tid >> 3;             // 0..31   (A: 8 passes of 32 rows)
  const int aK = (tid & 7) * 4;        // 0..28   (A: float4 along K)
  const int bKnn = tid >> 4;           // 0..15   (B NN: 2 passes of 16 k-rows)
  const int bNnn = (tid & 15) * 4;     // 0..60   (B NN: float4 along N)
  const int bNnt = tid >> 3;           // 0..31   (B NT: 2 passes of 32 n-rows)
  const int bKnt = (tid & 7) * 4;      // 0..28   (B NT: float4 along K)

  v8f accs[2][4] = {};

  for (int kk = 0; kk < K; kk += BK) {
    __syncthreads();

    // ---- stage A tile (BM x BK): float4 loads, packed bf16 b64 stores ----
    #pragma unroll
    for (int p = 0; p < 8; ++p) {
      int m = p * 32 + aM;
      const float4 v4 = *reinterpret_cast<const float4*>(&A[(long long)(m0 + m) * lda + kk + aK]);
      uint2 pk; pk.x = pack2(v4.x, v4.y); pk.y = pack2(v4.z, v4.w);
      *reinterpret_cast<uint2*>(&As[m][aK]) = pk;
    }

    // ---- stage B tile into Bs[n][k] (pre-transposed for fragment loads) ----
    if (bTrans) {                       // B is N x K: float4 along K
      #pragma unroll
      for (int p = 0; p < 2; ++p) {
        int n = p * 32 + bNnt;
        const float4 v4 = *reinterpret_cast<const float4*>(&Bm[(long long)(n0 + n) * ldb + kk + bKnt]);
        uint2 pk; pk.x = pack2(v4.x, v4.y); pk.y = pack2(v4.z, v4.w);
        *reinterpret_cast<uint2*>(&Bs[n][bKnt]) = pk;
      }
    } else {                            // B is K x N: float4 along N, transpose into LDS
      #pragma unroll
      for (int p = 0; p < 2; ++p) {
        int k = p * 16 + bKnn;
        const float4 v4 = *reinterpret_cast<const float4*>(&Bm[(long long)(kk + k) * ldb + n0 + bNnn]);
        *reinterpret_cast<unsigned short*>(&Bs[bNnn + 0][k]) = bf1(v4.x);
        *reinterpret_cast<unsigned short*>(&Bs[bNnn + 1][k]) = bf1(v4.y);
        *reinterpret_cast<unsigned short*>(&Bs[bNnn + 2][k]) = bf1(v4.z);
        *reinterpret_cast<unsigned short*>(&Bs[bNnn + 3][k]) = bf1(v4.w);
      }
    }

    // uniform prefetch of next A K-slice (global_prefetch_b8)
    if (kk + 2 * BK <= K) {
      __builtin_prefetch(&A[(long long)(m0 + tid) * lda + kk + BK + ((tid & 1) * 16)], 0, 0);
    }
    __syncthreads();

    // ---- A fragments (two M-strips): 16-bit A 16x32 layout (ISA 7.12.2) ----
    v8bf alo0 = *reinterpret_cast<const v8bf*>(&As[arow][akoff]);
    v8bf ahi0 = *reinterpret_cast<const v8bf*>(&As[arow][16 + akoff]);
    v8bf alo1 = *reinterpret_cast<const v8bf*>(&As[arow + 128][akoff]);
    v8bf ahi1 = *reinterpret_cast<const v8bf*>(&As[arow + 128][16 + akoff]);
    v16bf afrag0 = __builtin_shufflevector(alo0, ahi0, 0,1,2,3,4,5,6,7,8,9,10,11,12,13,14,15);
    v16bf afrag1 = __builtin_shufflevector(alo1, ahi1, 0,1,2,3,4,5,6,7,8,9,10,11,12,13,14,15);

    // ---- load 4 B fragments once, reuse for both M-strips: 8 WMMAs ----
    v16bf bfrag[4];
    #pragma unroll
    for (int f = 0; f < 4; ++f) {
      const bf16* bp = &Bs[f * 16 + lmod][half * 8];    // K = half*8 + i, 16 + half*8 + i (ISA table)
      v8bf blo = *reinterpret_cast<const v8bf*>(bp);
      v8bf bhi = *reinterpret_cast<const v8bf*>(bp + 16);
      bfrag[f] = __builtin_shufflevector(blo, bhi, 0,1,2,3,4,5,6,7,8,9,10,11,12,13,14,15);
    }
    #pragma unroll
    for (int f = 0; f < 4; ++f) {
      accs[0][f] = __builtin_amdgcn_wmma_f32_16x16x32_bf16(
          false, afrag0, false, bfrag[f], (short)0, accs[0][f], false, false);
    }
    #pragma unroll
    for (int f = 0; f < 4; ++f) {
      accs[1][f] = __builtin_amdgcn_wmma_f32_16x16x32_bf16(
          false, afrag1, false, bfrag[f], (short)0, accs[1][f], false, false);
    }
  }

  // ---- epilogue: C/D f32 16x16 layout: VGPR r -> M = r + 8*half, N = lmod ----
  // scale + bias (register pass)
  #pragma unroll
  for (int f = 0; f < 4; ++f) {
    float bb = bias ? bias[n0 + f * 16 + lmod] : 0.0f;
    #pragma unroll
    for (int r = 0; r < 8; ++r) {
      accs[0][f][r] = accs[0][f][r] * alpha + bb;
      accs[1][f][r] = accs[1][f][r] * alpha + bb;
    }
  }
  if (epi == EPI_GELU) {               // single uniform branch for activation
    #pragma unroll
    for (int s = 0; s < 2; ++s) {
      #pragma unroll
      for (int f = 0; f < 4; ++f) {
        #pragma unroll
        for (int r = 0; r < 8; ++r) {
          float x = accs[s][f][r];
          float t = tanhf(0.7978845608028654f * (x + 0.044715f * x * x * x));
          accs[s][f][r] = 0.5f * x * (1.0f + t);
        }
      }
    }
  }
  const int mBase = m0 + wave * 16 + 8 * half;
  #pragma unroll
  for (int s = 0; s < 2; ++s) {
    #pragma unroll
    for (int f = 0; f < 4; ++f) {
      int col = n0 + f * 16 + lmod;
      #pragma unroll
      for (int r = 0; r < 8; ++r) {
        long long row = mBase + s * 128 + r;
        float val = accs[s][f][r];
        if (Res) val += Res[row * ldr + col];
        C[row * ldc + col] = val;
      }
    }
  }
}

// x (B,C,384,384) -> xp (B,576,768), e = (py*16+px)*3 + c, p = hp*24 + wp
__global__ __launch_bounds__(256)
void patchify_kernel(const float* __restrict__ x, float* __restrict__ xp, int total)
{
  int idx = blockIdx.x * 256 + threadIdx.x;
  if (idx >= total) return;
  int e = idx % PDIM;
  int p = (idx / PDIM) % NPATCH;
  int b = idx / (PDIM * NPATCH);
  int c  = e % 3;
  int rem = e / 3;
  int px = rem % 16, py = rem / 16;
  int hp = p / 24, wp = p % 24;
  int row = hp * 16 + py, col = wp * 16 + px;
  xp[idx] = x[(((long long)b * 3 + c) * 384 + row) * 384 + col];
}

// t[b,l,:] = (l < 576 ? pe[b,l,:] : cls) + pos[l,:]   (real rows only)
__global__ __launch_bounds__(256)
void embed_kernel(const float* __restrict__ pe, const float* __restrict__ cls,
                  const float* __restrict__ pos, float* __restrict__ t, int total)
{
  int idx = blockIdx.x * 256 + threadIdx.x;
  if (idx >= total) return;
  int h = idx % HIDN;
  int l = (idx / HIDN) % LTOK;
  int b = idx / (HIDN * LTOK);
  float v = (l < NPATCH) ? pe[((long long)b * NPATCH + l) * HIDN + h] : cls[h];
  t[idx] = v + pos[(long long)l * HIDN + h];
}

// LayerNorm over 384 features; one row per block of 128 threads.
__global__ __launch_bounds__(128)
void layernorm_kernel(const float* __restrict__ X, long long xStride,
                      const float* __restrict__ g, const float* __restrict__ b,
                      float* __restrict__ Y, long long yStride)
{
  __shared__ float s1[128], s2[128];
  long long r = blockIdx.x;
  const float* x = X + r * xStride;
  int tid = threadIdx.x;
  float a0 = x[tid], a1 = x[tid + 128], a2 = x[tid + 256];
  s1[tid] = a0 + a1 + a2;
  s2[tid] = a0 * a0 + a1 * a1 + a2 * a2;
  __syncthreads();
  for (int s = 64; s > 0; s >>= 1) {
    if (tid < s) { s1[tid] += s1[tid + s]; s2[tid] += s2[tid + s]; }
    __syncthreads();
  }
  float mean = s1[0] * (1.0f / 384.0f);
  float var  = s2[0] * (1.0f / 384.0f) - mean * mean;
  float rstd = rsqrtf(var + 1e-5f);
  float* y = Y + r * yStride;
  y[tid]       = (a0 - mean) * rstd * g[tid]       + b[tid];
  y[tid + 128] = (a1 - mean) * rstd * g[tid + 128] + b[tid + 128];
  y[tid + 256] = (a2 - mean) * rstd * g[tid + 256] + b[tid + 256];
}

// softmax over L real elements of row (z, l); rows are LPAD wide, MPAD rows/z.
// Zero-fills the K-pad columns [L, LPAD) so the padded AV GEMM is exact.
__global__ __launch_bounds__(256)
void softmax_kernel(float* __restrict__ S, int L, int LPAD, int MPAD)
{
  __shared__ float red[256];
  long long z = blockIdx.y, l = blockIdx.x;
  float* p = S + (z * MPAD + l) * LPAD;
  int tid = threadIdx.x;
  float mx = -3.4e38f;
  for (int i = tid; i < L; i += 256) mx = fmaxf(mx, p[i]);
  red[tid] = mx; __syncthreads();
  for (int s = 128; s > 0; s >>= 1) {
    if (tid < s) red[tid] = fmaxf(red[tid], red[tid + s]);
    __syncthreads();
  }
  mx = red[0]; __syncthreads();
  float sum = 0.0f;
  for (int i = tid; i < L; i += 256) { float e = __expf(p[i] - mx); p[i] = e; sum += e; }
  red[tid] = sum; __syncthreads();
  for (int s = 128; s > 0; s >>= 1) {
    if (tid < s) red[tid] += red[tid + s];
    __syncthreads();
  }
  float inv = 1.0f / red[0];
  for (int i = tid; i < L; i += 256) p[i] *= inv;
  for (int i = L + tid; i < LPAD; i += 256) p[i] = 0.0f;
}

// (B*H, L_MPAD, 64) -> (B*H, LTOK, 64) contiguous == reference's scrambled flat view
__global__ __launch_bounds__(256)
void repack_av_kernel(const float* __restrict__ src, float* __restrict__ dst, int total)
{
  int idx = blockIdx.x * 256 + threadIdx.x;
  if (idx >= total) return;
  int d = idx & 63;
  int l = (idx >> 6) % LTOK;
  int z = idx / (64 * LTOK);
  dst[idx] = src[((long long)z * L_MPAD + l) * 64 + d];
}

// zero-padded copies of head_W (384x1000 -> 384x1024) and head_b (1000 -> 1024)
__global__ __launch_bounds__(256)
void pad_head_kernel(const float* __restrict__ W, const float* __restrict__ b,
                     float* __restrict__ Wp, float* __restrict__ bp, int total)
{
  int idx = blockIdx.x * 256 + threadIdx.x;
  if (idx >= total) return;
  int row = idx >> 10, col = idx & 1023;
  if (row < HIDN) Wp[idx] = (col < OUT_N) ? W[(long long)row * OUT_N + col] : 0.0f;
  else            bp[col] = (col < OUT_N) ? b[col] : 0.0f;
}

// head_tmp (256x1024) -> out (16x1000)
__global__ __launch_bounds__(256)
void copy_out_kernel(const float* __restrict__ src, float* __restrict__ dst, int total)
{
  int idx = blockIdx.x * 256 + threadIdx.x;
  if (idx >= total) return;
  int r = idx / OUT_N, c = idx % OUT_N;
  dst[idx] = src[(long long)r * OUT_PAD + c];
}

extern "C" void kernel_launch(void* const* d_in, const int* in_sizes, int n_in,
                              void* d_out, int out_size, void* d_ws, size_t ws_size,
                              hipStream_t stream) {
  (void)in_sizes; (void)n_in; (void)out_size; (void)ws_size;
  const float* x       = (const float*)d_in[0];
  const float* patch_W = (const float*)d_in[1];
  const float* patch_b = (const float*)d_in[2];
  const float* cls     = (const float*)d_in[3];
  const float* pos     = (const float*)d_in[4];
  const float* ln1_g   = (const float*)d_in[5];
  const float* ln1_b   = (const float*)d_in[6];
  const float* Wq      = (const float*)d_in[7];
  const float* bq      = (const float*)d_in[8];
  const float* Wk      = (const float*)d_in[9];
  const float* bk      = (const float*)d_in[10];
  const float* Wv      = (const float*)d_in[11];
  const float* bv      = (const float*)d_in[12];
  const float* Wo      = (const float*)d_in[13];
  const float* bo      = (const float*)d_in[14];
  const float* ln2_g   = (const float*)d_in[15];
  const float* ln2_b   = (const float*)d_in[16];
  const float* W1      = (const float*)d_in[17];
  const float* b1      = (const float*)d_in[18];
  const float* W2      = (const float*)d_in[19];
  const float* b2      = (const float*)d_in[20];
  const float* lnf_g   = (const float*)d_in[21];
  const float* lnf_b   = (const float*)d_in[22];
  const float* head_W  = (const float*)d_in[23];
  const float* head_b  = (const float*)d_in[24];
  float* out = (float*)d_out;

  // --- workspace carve-out (~485 MB; padded scores dominate).
  //     Pad rows/cols only ever feed pad rows/cols (softmax zeroes pad K). ---
  size_t off = 0;
  auto carve = [&](size_t bytes) -> float* {
    float* p = (float*)((char*)d_ws + off);
    off += (bytes + 255) & ~(size_t)255;
    return p;
  };
  const int ZB = BATCH_N * HEADS_N;                       // 128 (b,h) pairs
  float* xp  = carve((size_t)BATCH_N * NPATCH * PDIM   * 4);
  float* pe  = carve((size_t)BATCH_N * NPATCH * HIDN   * 4);
  float* t   = carve((size_t)MROWS_PAD * HIDN   * 4);
  float* y   = carve((size_t)MROWS_PAD * HIDN   * 4);
  float* qb  = carve((size_t)MROWS_PAD * INNER_N * 4);
  float* kb  = carve((size_t)MROWS_PAD * INNER_N * 4);
  float* vb  = carve((size_t)MROWS_PAD * INNER_N * 4);
  float* avp = carve((size_t)ZB * L_MPAD * DH_N * 4);     // padded (B,H,768,64)
  float* avf = carve((size_t)MROWS_PAD * INNER_N * 4);    // packed flat (B*L,512)
  float* sc  = carve((size_t)ZB * L_MPAD * L_PAD * 4);    // (B*H,768,640)
  float* h1  = carve((size_t)MROWS_PAD * MLP_N  * 4);
  float* lnf = carve((size_t)256 * HIDN * 4);
  float* hwp = carve((size_t)HIDN * OUT_PAD * 4);
  float* hbp = carve((size_t)OUT_PAD * 4);
  float* hto = carve((size_t)256 * OUT_PAD * 4);

  auto gemm = [&](const float* A, long long lda, long long aO, long long aI,
                  const float* Bp, long long ldb, long long bO, long long bI, int bT,
                  const float* bias, const float* Res, long long ldr, long long rO, long long rI,
                  float* Cp, long long ldc, long long cO, long long cI,
                  int Mp, int Np, int K, int batch, int div, float alpha, int epi) {
    dim3 grid(Np / BN, Mp / BM, batch);
    gemm_wmma_bf16<<<grid, dim3(256), 0, stream>>>(
        A, lda, aO, aI, Bp, ldb, bO, bI, bT, bias, Res, ldr, rO, rI,
        Cp, ldc, cO, cI, K, div, alpha, epi);
  };

  const float scale = 0.125f;      // 1/sqrt(64)

  // 1) patchify + patch embedding + token/pos assembly
  {
    int total = BATCH_N * NPATCH * PDIM;
    patchify_kernel<<<(total + 255) / 256, 256, 0, stream>>>(x, xp, total);
  }
  gemm(xp, PDIM, 0, 0, patch_W, HIDN, 0, 0, 0, patch_b,
       nullptr, 0, 0, 0, pe, HIDN, 0, 0,
       BATCH_N * NPATCH, HIDN, PDIM, 1, 1, 1.0f, EPI_NONE);   // 9216 = 36*256 exact
  {
    int total = MROWS * HIDN;
    embed_kernel<<<(total + 255) / 256, 256, 0, stream>>>(pe, cls, pos, t, total);
  }

  // 2) transformer layers
  const long long LSC = (long long)L_MPAD * L_PAD; // per-(b,h) score stride
  const long long LI  = (long long)LTOK * INNER_N; // per-b stride in q/k/v views
  const long long LPD = (long long)L_MPAD * DH_N;  // per-(b,h) stride in avp
  for (int d = 0; d < DEPTH_N; ++d) {
    const float* wq = Wq + (long long)d * HIDN * INNER_N;
    const float* wk = Wk + (long long)d * HIDN * INNER_N;
    const float* wv = Wv + (long long)d * HIDN * INNER_N;
    const float* wo = Wo + (long long)d * INNER_N * HIDN;
    const float* w1 = W1 + (long long)d * HIDN * MLP_N;
    const float* w2 = W2 + (long long)d * MLP_N * HIDN;

    // LN1: t -> y (real rows only; pad rows stay garbage and feed only pad rows)
    layernorm_kernel<<<MROWS, 128, 0, stream>>>(t, HIDN, ln1_g + d * HIDN, ln1_b + d * HIDN, y, HIDN);

    // q/k/v = y @ W + b   (M=9472 pad, K=384, N=512)
    gemm(y, HIDN, 0, 0, wq, INNER_N, 0, 0, 0, bq + d * INNER_N,
         nullptr, 0, 0, 0, qb, INNER_N, 0, 0, MROWS_PAD, INNER_N, HIDN, 1, 1, 1.0f, EPI_NONE);
    gemm(y, HIDN, 0, 0, wk, INNER_N, 0, 0, 0, bk + d * INNER_N,
         nullptr, 0, 0, 0, kb, INNER_N, 0, 0, MROWS_PAD, INNER_N, HIDN, 1, 1, 1.0f, EPI_NONE);
    gemm(y, HIDN, 0, 0, wv, INNER_N, 0, 0, 0, bv + d * INNER_N,
         nullptr, 0, 0, 0, vb, INNER_N, 0, 0, MROWS_PAD, INNER_N, HIDN, 1, 1, 1.0f, EPI_NONE);

    // scores[z=(b,h)] = scale * Q_bh @ K_bh^T   (M=768, N=640, K=64, NT)
    gemm(qb, INNER_N, LI, DH_N,
         kb, INNER_N, LI, DH_N, /*bTrans=*/1,
         nullptr, nullptr, 0, 0, 0,
         sc, L_PAD, (long long)HEADS_N * LSC, LSC,
         L_MPAD, L_PAD, DH_N, ZB, HEADS_N, scale, EPI_NONE);

    // softmax over 577 real cols of the 577 real rows; zero pad cols
    softmax_kernel<<<dim3(LTOK, ZB), 256, 0, stream>>>(sc, LTOK, L_PAD, L_MPAD);

    // av_pad[z] = P_bh @ V_bh  (M=768, N=64, K=640; pad-K is exact zeros)
    gemm(sc, L_PAD, (long long)HEADS_N * LSC, LSC,
         vb, INNER_N, LI, DH_N, /*bTrans=*/0,
         nullptr, nullptr, 0, 0, 0,
         avp, DH_N, (long long)HEADS_N * LPD, LPD,
         L_MPAD, DH_N, L_PAD, ZB, HEADS_N, 1.0f, EPI_NONE);

    // repack (B,H,768,64) -> (B,H,577,64) contiguous == reference reshape view
    {
      int total = ZB * LTOK * DH_N;
      repack_av_kernel<<<(total + 255) / 256, 256, 0, stream>>>(avp, avf, total);
    }

    // t = t + av_flat @ Wo + bo
    gemm(avf, INNER_N, 0, 0, wo, HIDN, 0, 0, 0, bo + d * HIDN,
         t, HIDN, 0, 0, t, HIDN, 0, 0, MROWS_PAD, HIDN, INNER_N, 1, 1, 1.0f, EPI_NONE);

    // LN2: t -> y
    layernorm_kernel<<<MROWS, 128, 0, stream>>>(t, HIDN, ln2_g + d * HIDN, ln2_b + d * HIDN, y, HIDN);

    // h1 = gelu(y @ W1 + b1)
    gemm(y, HIDN, 0, 0, w1, MLP_N, 0, 0, 0, b1 + d * MLP_N,
         nullptr, 0, 0, 0, h1, MLP_N, 0, 0, MROWS_PAD, MLP_N, HIDN, 1, 1, 1.0f, EPI_GELU);

    // t = t + h1 @ W2 + b2
    gemm(h1, MLP_N, 0, 0, w2, HIDN, 0, 0, 0, b2 + d * HIDN,
         t, HIDN, 0, 0, t, HIDN, 0, 0, MROWS_PAD, HIDN, MLP_N, 1, 1, 1.0f, EPI_NONE);
  }

  // 3) pooled = t[:,0]; final LN (strided row read) + padded head GEMM + copy out
  layernorm_kernel<<<BATCH_N, 128, 0, stream>>>(t, (long long)LTOK * HIDN, lnf_g, lnf_b, lnf, HIDN);
  {
    int total = (HIDN + 1) * OUT_PAD;
    pad_head_kernel<<<(total + 255) / 256, 256, 0, stream>>>(head_W, head_b, hwp, hbp, total);
  }
  gemm(lnf, HIDN, 0, 0, hwp, OUT_PAD, 0, 0, 0, hbp,
       nullptr, 0, 0, 0, hto, OUT_PAD, 0, 0, 256, OUT_PAD, HIDN, 1, 1, 1.0f, EPI_NONE);
  {
    int total = BATCH_N * OUT_N;
    copy_out_kernel<<<(total + 255) / 256, 256, 0, stream>>>(hto, out, total);
  }
}
